// Multi_Query_Attention_68564857914085
// MI455X (gfx1250) — matrix operand order repaired and checked by reading, not compiled, not run on hardware
//
#include <hip/hip_runtime.h>

// ---------------------------------------------------------------------------
// MQA for MI455X (gfx1250): bf16 WMMA, f32 accumulation, async-LDS staging.
// B=2, S=2048, D=1024, H=16, DH=64
// ---------------------------------------------------------------------------

typedef unsigned short u16;
typedef __attribute__((ext_vector_type(4)))  u16    u16x4;
typedef __attribute__((ext_vector_type(8)))  u16    u16x8;
typedef __attribute__((ext_vector_type(16))) u16    u16x16;
typedef __attribute__((ext_vector_type(16))) __bf16 bf16x16;
typedef __attribute__((ext_vector_type(8)))  float  f32x8;

static constexpr int kB  = 2;
static constexpr int kS  = 2048;
static constexpr int kD  = 1024;
static constexpr int kH  = 16;
static constexpr int kDH = 64;
static constexpr int kM  = kB * kS;   // 4096 rows for the projection GEMMs

__device__ __forceinline__ u16 f2bf(float f) {
    unsigned u = __builtin_bit_cast(unsigned, f);
    u += 0x7fffu + ((u >> 16) & 1u);          // round-to-nearest-even
    return (u16)(u >> 16);
}

__device__ __forceinline__ bf16x16 mkfrag(const u16* lo, const u16* hi) {
    u16x8 l = *(const u16x8*)lo;
    u16x8 h = *(const u16x8*)hi;
    u16x16 u = __builtin_shufflevector(l, h, 0,1,2,3,4,5,6,7,8,9,10,11,12,13,14,15);
    return __builtin_bit_cast(bf16x16, u);
}

__device__ __forceinline__ f32x8 wmma_bf16(bf16x16 a, bf16x16 b, f32x8 c) {
    return __builtin_amdgcn_wmma_f32_16x16x32_bf16(
        false, a, false, b, (short)0, c, false, false);
}

// 16-byte async copy global -> LDS (ASYNCcnt-tracked DMA, bypasses VGPRs)
__device__ __forceinline__ void async_cp16(const u16* gptr, u16* lptr) {
    unsigned           lds = (unsigned)(unsigned long long)lptr; // addr[31:0] = LDS offset
    unsigned long long ga  = (unsigned long long)gptr;
    asm volatile("global_load_async_to_lds_b128 %0, %1, off"
                 :: "v"(lds), "v"(ga) : "memory");
}

// --------------------------- elementwise converts --------------------------

__global__ __launch_bounds__(256) void cvt_f32_bf16_x4(
        const float* __restrict__ in, u16* __restrict__ out, int n4) {
    int i = blockIdx.x * blockDim.x + threadIdx.x;
    if (i >= n4) return;
    float4 v = ((const float4*)in)[i];
    u16x4 o = { f2bf(v.x), f2bf(v.y), f2bf(v.z), f2bf(v.w) };
    *(u16x4*)(out + 4 * i) = o;
}

// Wt[n*K + k] = bf16(W[k*N + n])   (W is K x N row-major)
__global__ __launch_bounds__(256) void transpose_cvt_w(
        const float* __restrict__ W, u16* __restrict__ Wt, int K, int N) {
    int i = blockIdx.x * blockDim.x + threadIdx.x;
    if (i >= K * N) return;
    int k = i / N, n = i - k * N;
    Wt[(size_t)n * K + k] = f2bf(W[i]);
}

// Vt[b][dh][s] = V[b][s][dh]   (bf16 -> bf16)
__global__ __launch_bounds__(256) void transpose_v(
        const u16* __restrict__ V, u16* __restrict__ Vt) {
    int i = blockIdx.x * blockDim.x + threadIdx.x;     // i = b*S*DH + s*DH + dh
    if (i >= kB * kS * kDH) return;
    int dh = i & (kDH - 1);
    int s  = (i / kDH) & (kS - 1);
    int b  = i / (kDH * kS);
    Vt[((size_t)b * kDH + dh) * kS + s] = V[i];
}

// ------------------------------ WMMA GEMM ----------------------------------
// C(M,N) = A(M,K)bf16 @ Wt(N,K)bf16^T + bias ; one wave -> 32x64 tile
// (2 A-fragments reuse each 4 B-fragments: 8 WMMAs per 6KB of fragments).
__global__ __launch_bounds__(256) void gemm_bf16_tn(
        const u16* __restrict__ A, const u16* __restrict__ Wt,
        const float* __restrict__ bias,
        float* __restrict__ outf, u16* __restrict__ outb,
        int M, int N, int K) {
    int wave = (blockIdx.x * blockDim.x + threadIdx.x) >> 5;
    int lane = threadIdx.x & 31;
    int half = lane >> 4;
    int l16  = lane & 15;

    int mtiles = M >> 5;                  // 32-row tiles
    int ngrps  = N >> 6;                  // 64-col groups
    if (wave >= mtiles * ngrps) return;
    int mt = wave % mtiles;
    int ng = wave / mtiles;

    const u16* arow0 = A + (size_t)(mt * 32 +  0 + l16) * K + half * 16;
    const u16* arow1 = A + (size_t)(mt * 32 + 16 + l16) * K + half * 16;
    const u16* brow0 = Wt + (size_t)(ng * 64 +  0 + l16) * K + half * 16;
    const u16* brow1 = Wt + (size_t)(ng * 64 + 16 + l16) * K + half * 16;
    const u16* brow2 = Wt + (size_t)(ng * 64 + 32 + l16) * K + half * 16;
    const u16* brow3 = Wt + (size_t)(ng * 64 + 48 + l16) * K + half * 16;

    f32x8 acc[2][4] = {};
    for (int kb = 0; kb < K; kb += 32) {
        bf16x16 a0 = mkfrag(arow0 + kb, arow0 + kb + 8);
        bf16x16 a1 = mkfrag(arow1 + kb, arow1 + kb + 8);
        bf16x16 b0 = mkfrag(brow0 + kb, brow0 + kb + 8);
        bf16x16 b1 = mkfrag(brow1 + kb, brow1 + kb + 8);
        bf16x16 b2 = mkfrag(brow2 + kb, brow2 + kb + 8);
        bf16x16 b3 = mkfrag(brow3 + kb, brow3 + kb + 8);
        acc[0][0] = wmma_bf16(a0, b0, acc[0][0]);
        acc[0][1] = wmma_bf16(a0, b1, acc[0][1]);
        acc[0][2] = wmma_bf16(a0, b2, acc[0][2]);
        acc[0][3] = wmma_bf16(a0, b3, acc[0][3]);
        acc[1][0] = wmma_bf16(a1, b0, acc[1][0]);
        acc[1][1] = wmma_bf16(a1, b1, acc[1][1]);
        acc[1][2] = wmma_bf16(a1, b2, acc[1][2]);
        acc[1][3] = wmma_bf16(a1, b3, acc[1][3]);
    }

#pragma unroll
    for (int s = 0; s < 2; ++s) {
#pragma unroll
        for (int t = 0; t < 4; ++t) {
            int col = ng * 64 + t * 16 + l16;
            float bv = bias[col];
#pragma unroll
            for (int r = 0; r < 8; ++r) {
                float v = acc[s][t][r] + bv;
                size_t off = (size_t)(mt * 32 + s * 16 + r + half * 8) * N + col;
                if (outf) outf[off] = v;
                else      outb[off] = f2bf(v);
            }
        }
    }
}

// --------------------------- flash attention -------------------------------
// Block = 8 waves sharing one (b,h); each wave owns 16 query rows.
// K/V tiles (32 keys) staged once per block in LDS via async DMA,
// double-buffered; all fragment loads for QK^T and PV come from LDS.
__global__ __launch_bounds__(256) void flash_mqa(
        const u16* __restrict__ Q,   // (B*S, D) bf16, head h at col h*64
        const u16* __restrict__ K,   // (B*S, DH) bf16
        const u16* __restrict__ Vt,  // (B, DH, S) bf16
        u16* __restrict__ attn) {    // (B*S, D) bf16
    // padded rows -> 16B-aligned b128 ds loads, bank-conflict-free
    __shared__ alignas(16) u16 kbuf[2][32][72];   // 32 keys x 64 dh (144B rows)
    __shared__ alignas(16) u16 vbuf[2][64][40];   // 64 dh x 32 keys (80B rows)
    __shared__ alignas(16) u16 pbuf[8][16][40];   // per-wave P tile (80B rows)

    int tid  = threadIdx.x;
    int wid  = tid >> 5;
    int lane = tid & 31;
    int half = lane >> 4;
    int l16  = lane & 15;

    int gw = blockIdx.x * 8 + wid;                // 4096 waves total
    int mt = gw & 127;                            // S/16 = 128 (varies per wave)
    int h  = (gw >> 7) & 15;                      // shared by the block
    int b  = gw >> 11;                            // shared by the block

    const float scale = 0.125f;                   // 1/sqrt(64)

    // Q fragments (k-dim 0..31 and 32..63) held in registers
    const u16* qp = Q + (size_t)(b * kS + mt * 16 + l16) * kD + h * kDH + half * 16;
    bf16x16 aq0 = mkfrag(qp +  0, qp +  8);
    bf16x16 aq1 = mkfrag(qp + 32, qp + 40);

    const u16* Kb  = K  + (size_t)b * kS * kDH;
    const u16* Vtb = Vt + (size_t)b * kDH * kS;

    // per-thread copy slots (256 threads move 4KB K + 4KB V per tile)
    int kkey = tid >> 3, kpart = tid & 7;         // K: 32 rows x 8 chunks
    int vdh  = tid >> 2, vpart = tid & 3;         // V: 64 rows x 4 chunks

    float mrow[8], lrow[8];
#pragma unroll
    for (int r = 0; r < 8; ++r) { mrow[r] = -3.0e38f; lrow[r] = 0.0f; }
    f32x8 o0 = {}, o1 = {}, o2 = {}, o3 = {};

    constexpr int J = kS / 32;                    // 64 key tiles

    // preload tile 0 into buffer 0
    async_cp16(Kb  + (size_t)kkey * kDH + kpart * 8, &kbuf[0][kkey][kpart * 8]);
    async_cp16(Vtb + (size_t)vdh  * kS  + vpart * 8, &vbuf[0][vdh][vpart * 8]);

    for (int j = 0; j < J; ++j) {
        int cb = j & 1, nb = cb ^ 1;
        if (j + 1 < J) {
            int kn = (j + 1) * 32;
            async_cp16(Kb  + (size_t)(kn + kkey) * kDH + kpart * 8,
                       &kbuf[nb][kkey][kpart * 8]);
            async_cp16(Vtb + (size_t)vdh * kS + kn + vpart * 8,
                       &vbuf[nb][vdh][vpart * 8]);
            // in-order completion: <=2 outstanding means tile j has landed
            asm volatile("s_wait_asynccnt 0x2" ::: "memory");
        } else {
            asm volatile("s_wait_asynccnt 0x0" ::: "memory");
        }
        __syncthreads();                          // tile j visible to all waves

        // ---- scores: two 16x16 tiles from LDS K tile ----
        f32x8 c0 = {}, c1 = {};
        {
            const u16* kp0a = &kbuf[cb][l16][ 0 + half * 16];
            const u16* kp0b = &kbuf[cb][l16][32 + half * 16];
            const u16* kp1a = &kbuf[cb][16 + l16][ 0 + half * 16];
            const u16* kp1b = &kbuf[cb][16 + l16][32 + half * 16];
            c0 = wmma_bf16(aq0, mkfrag(kp0a, kp0a + 8), c0);
            c0 = wmma_bf16(aq1, mkfrag(kp0b, kp0b + 8), c0);
            c1 = wmma_bf16(aq0, mkfrag(kp1a, kp1a + 8), c1);
            c1 = wmma_bf16(aq1, mkfrag(kp1b, kp1b + 8), c1);
        }

        // ---- online softmax (row stats replicated per half-wave) ----
        float p0[8], p1[8];
#pragma unroll
        for (int r = 0; r < 8; ++r) {
            float s0 = c0[r] * scale;
            float s1 = c1[r] * scale;
            float tmax = fmaxf(s0, s1);
#pragma unroll
            for (int msk = 1; msk < 16; msk <<= 1)
                tmax = fmaxf(tmax, __shfl_xor(tmax, msk, 32));
            float nm = fmaxf(mrow[r], tmax);
            float e0 = __expf(s0 - nm);
            float e1 = __expf(s1 - nm);
            float rs = e0 + e1;
#pragma unroll
            for (int msk = 1; msk < 16; msk <<= 1)
                rs += __shfl_xor(rs, msk, 32);
            float corr = __expf(mrow[r] - nm);
            lrow[r] = lrow[r] * corr + rs;
            mrow[r] = nm;
            p0[r] = e0; p1[r] = e1;
            o0[r] *= corr; o1[r] *= corr; o2[r] *= corr; o3[r] *= corr;
        }

        // ---- reshape P (C layout -> A layout) via padded LDS tile ----
        u16 (*pp)[40] = pbuf[wid];
#pragma unroll
        for (int r = 0; r < 8; ++r) {
            int row = r + half * 8;
            pp[row][l16]      = f2bf(p0[r]);
            pp[row][16 + l16] = f2bf(p1[r]);
        }
        asm volatile("s_wait_dscnt 0x0" ::: "memory");
        bf16x16 pa = mkfrag(&pp[l16][half * 16], &pp[l16][half * 16 + 8]);

        // ---- O += P (16x32) x V tile (32x64) from LDS V tile ----
        {
            const u16* vp0 = &vbuf[cb][ 0 + l16][half * 16];
            const u16* vp1 = &vbuf[cb][16 + l16][half * 16];
            const u16* vp2 = &vbuf[cb][32 + l16][half * 16];
            const u16* vp3 = &vbuf[cb][48 + l16][half * 16];
            o0 = wmma_bf16(pa, mkfrag(vp0, vp0 + 8), o0);
            o1 = wmma_bf16(pa, mkfrag(vp1, vp1 + 8), o1);
            o2 = wmma_bf16(pa, mkfrag(vp2, vp2 + 8), o2);
            o3 = wmma_bf16(pa, mkfrag(vp3, vp3 + 8), o3);
        }

        __syncthreads();     // all waves done reading tile j before it is overwritten
    }

    // ---- normalize + write attn (B,S,H,DH) bf16 ----
    f32x8 oo[4] = {o0, o1, o2, o3};
#pragma unroll
    for (int r = 0; r < 8; ++r) {
        float inv = 1.0f / lrow[r];
        size_t row = (size_t)(b * kS + mt * 16 + r + half * 8) * kD + h * kDH;
#pragma unroll
        for (int t = 0; t < 4; ++t)
            attn[row + t * 16 + l16] = f2bf(oo[t][r] * inv);
    }
}

// ------------------------------ launcher -----------------------------------

extern "C" void kernel_launch(void* const* d_in, const int* in_sizes, int n_in,
                              void* d_out, int out_size, void* d_ws, size_t ws_size,
                              hipStream_t stream) {
    const float* query = (const float*)d_in[0];
    const float* value = (const float*)d_in[1];
    const float* Wq    = (const float*)d_in[2];
    const float* bq    = (const float*)d_in[3];
    const float* Wk    = (const float*)d_in[4];
    const float* bk    = (const float*)d_in[5];
    const float* Wv    = (const float*)d_in[6];
    const float* bv    = (const float*)d_in[7];
    const float* Wo    = (const float*)d_in[8];
    const float* bo    = (const float*)d_in[9];
    float* out = (float*)d_out;

    size_t off = 0;
    auto carve = [&](size_t bytes) {
        size_t r = off;
        off += (bytes + 255) & ~(size_t)255;
        return r;
    };
    char* W = (char*)d_ws;
    const size_t BSD  = (size_t)kB * kS * kD  * sizeof(u16);   // 8 MB
    const size_t BSDH = (size_t)kB * kS * kDH * sizeof(u16);   // 512 KB
    u16* qbf  = (u16*)(W + carve(BSD));
    u16* vbf  = (u16*)(W + carve(BSD));
    u16* Qb   = (u16*)(W + carve(BSD));
    u16* attn = (u16*)(W + carve(BSD));
    u16* Kb   = (u16*)(W + carve(BSDH));
    u16* Vb   = (u16*)(W + carve(BSDH));
    u16* Vt   = (u16*)(W + carve(BSDH));
    u16* WqT  = (u16*)(W + carve((size_t)kD * kD  * sizeof(u16)));
    u16* WoT  = (u16*)(W + carve((size_t)kD * kD  * sizeof(u16)));
    u16* WkT  = (u16*)(W + carve((size_t)kD * kDH * sizeof(u16)));
    u16* WvT  = (u16*)(W + carve((size_t)kD * kDH * sizeof(u16)));
    if (off > ws_size) return;   // ~40 MB required

    const int n4 = kB * kS * kD / 4;                       // 1,048,576
    cvt_f32_bf16_x4<<<n4 / 256, 256, 0, stream>>>(query, qbf, n4);
    cvt_f32_bf16_x4<<<n4 / 256, 256, 0, stream>>>(value, vbf, n4);

    transpose_cvt_w<<<(kD * kD ) / 256, 256, 0, stream>>>(Wq, WqT, kD, kD);
    transpose_cvt_w<<<(kD * kD ) / 256, 256, 0, stream>>>(Wo, WoT, kD, kD);
    transpose_cvt_w<<<(kD * kDH) / 256, 256, 0, stream>>>(Wk, WkT, kD, kDH);
    transpose_cvt_w<<<(kD * kDH) / 256, 256, 0, stream>>>(Wv, WvT, kD, kDH);

    // Q projection: (4096,1024)x(1024,1024) -> bf16   (32x64 tile/wave)
    gemm_bf16_tn<<<(kM / 32) * (kD / 64) / 8, 256, 0, stream>>>(
        qbf, WqT, bq, nullptr, Qb, kM, kD, kD);
    // K,V projections: (4096,1024)x(1024,64) -> bf16
    gemm_bf16_tn<<<(kM / 32) * (kDH / 64) / 8, 256, 0, stream>>>(
        vbf, WkT, bk, nullptr, Kb, kM, kDH, kD);
    gemm_bf16_tn<<<(kM / 32) * (kDH / 64) / 8, 256, 0, stream>>>(
        vbf, WvT, bv, nullptr, Vb, kM, kDH, kD);

    transpose_v<<<(kB * kS * kDH) / 256, 256, 0, stream>>>(Vb, Vt);

    // flash attention: B*H*(S/16) = 4096 waves, 8 waves/block share (b,h)
    flash_mqa<<<(kB * kH * (kS / 16)) / 8, 256, 0, stream>>>(Qb, Kb, Vt, attn);

    // output projection -> f32 d_out
    gemm_bf16_tn<<<(kM / 32) * (kD / 64) / 8, 256, 0, stream>>>(
        attn, WoT, bo, out, nullptr, kM, kD, kD);
}
